// GAT_40896678592681
// MI455X (gfx1250) — compile-verified, matrix operand order checked by reading, not run on hardware
//
#include <hip/hip_runtime.h>
#include <math.h>

#define IN_DIM 256
#define NHID   32
#define HEADS  4
#define HID    (HEADS * NHID)   // 128
#define OUT_DIM 128
#define NEG_SLOPE 0.2f

typedef __attribute__((ext_vector_type(16))) __bf16 v16bf;
typedef __attribute__((ext_vector_type(8)))  __bf16 v8bf;
typedef __attribute__((ext_vector_type(8)))  float  v8f;

// ---------- ordered-uint encoding for float atomicMax (handles negatives) ----------
__device__ __forceinline__ unsigned f2ord(float f) {
    unsigned b = __float_as_uint(f);
    return (b & 0x80000000u) ? ~b : (b | 0x80000000u);
}
__device__ __forceinline__ float ord2f(unsigned k) {
    unsigned b = (k & 0x80000000u) ? (k & 0x7FFFFFFFu) : ~k;
    return __uint_as_float(b);
}
#define ORD_NEG_MAX 0x00800000u   // f2ord(-FLT_MAX)

// ---------- one-time f32 -> bf16 hi/lo split (bf16x3 compensation) ----------
__global__ void split_bf16_kernel(const float* __restrict__ x, __bf16* __restrict__ hi,
                                  __bf16* __restrict__ lo, long long n) {
    long long i = blockIdx.x * (long long)blockDim.x + threadIdx.x;
    if (i >= n) return;
    float v = x[i];
    __bf16 h = (__bf16)v;
    hi[i] = h;
    lo[i] = (__bf16)(v - (float)h);
}

// f32 W[K,N] -> transposed bf16 hi/lo [N,K]  (so B fragments are row-contiguous)
__global__ void split_bf16_t_kernel(const float* __restrict__ Wm, __bf16* __restrict__ hi,
                                    __bf16* __restrict__ lo, int K, int N) {
    int i = blockIdx.x * blockDim.x + threadIdx.x;
    if (i >= K * N) return;
    int k = i / N, n = i % N;
    float v = Wm[i];
    __bf16 h = (__bf16)v;
    hi[(size_t)n * K + k] = h;
    lo[(size_t)n * K + k] = (__bf16)(v - (float)h);
}

// 16-bit fragment: lane's 16 elems are K = [base, base+8) ++ [base+16, base+24)
// -> two aligned 16B loads + shuffle.  (base = k0 + 8*g within a row)
__device__ __forceinline__ v16bf frag16(const __bf16* __restrict__ p) {
    v8bf lo = *(const v8bf*)p;
    v8bf hi = *(const v8bf*)(p + 16);
    return __builtin_shufflevector(lo, hi, 0, 1, 2, 3, 4, 5, 6, 7,
                                           8, 9, 10, 11, 12, 13, 14, 15);
}

__device__ __forceinline__ v8f wmma_bf16(v16bf a, v16bf b, v8f c) {
    return __builtin_amdgcn_wmma_f32_16x16x32_bf16(false, a, false, b, (short)0, c,
                                                   false, false);
}

// ---------- bf16x3 WMMA GEMM: C[M,N] = A[M,K] * B[K,N], A/B preconverted ----------
// Block = 256 thr = 8 waves. Wave w owns N-tile w (N=128); computes 32x16 (2 accum),
// reusing B fragments. K multiple of 32, N multiple of 128.
__global__ void __launch_bounds__(256)
gemm_wmma_bf16x3(const __bf16* __restrict__ Ah, const __bf16* __restrict__ Al,
                 const __bf16* __restrict__ Bth, const __bf16* __restrict__ Btl,
                 float* __restrict__ C, int M, int K, int N) {
    const int w  = threadIdx.x >> 5;            // wave -> N tile
    const int t  = threadIdx.x & 31;
    const int g  = t >> 4;                      // lane half
    const int ml = t & 15;                      // row (A) / col (B) in tile
    const int n0 = (blockIdx.y * 8 + w) * 16;
    const int m0 = blockIdx.x * 32;

    int r0 = m0 + ml;      r0 = r0 < M ? r0 : M - 1;   // clamp keeps EXEC uniform
    int r1 = m0 + 16 + ml; r1 = r1 < M ? r1 : M - 1;

    const __bf16* a0h = Ah + (size_t)r0 * K;
    const __bf16* a0l = Al + (size_t)r0 * K;
    const __bf16* a1h = Ah + (size_t)r1 * K;
    const __bf16* a1l = Al + (size_t)r1 * K;
    const __bf16* bh  = Bth + (size_t)(n0 + ml) * K;
    const __bf16* bl  = Btl + (size_t)(n0 + ml) * K;

    v8f acc0 = {}, acc1 = {};
    for (int k0 = 0; k0 < K; k0 += 32) {
        int o = k0 + 8 * g;
        v16bf fbh = frag16(bh + o);
        v16bf fbl = frag16(bl + o);
        v16bf f0h = frag16(a0h + o);
        v16bf f0l = frag16(a0l + o);
        v16bf f1h = frag16(a1h + o);
        v16bf f1l = frag16(a1l + o);
        // a*b ~= ah*bh + ah*bl + al*bh (al*bl ~ 2^-32 dropped), f32 accumulate
        acc0 = wmma_bf16(f0h, fbh, acc0);
        acc0 = wmma_bf16(f0h, fbl, acc0);
        acc0 = wmma_bf16(f0l, fbh, acc0);
        acc1 = wmma_bf16(f1h, fbh, acc1);
        acc1 = wmma_bf16(f1h, fbl, acc1);
        acc1 = wmma_bf16(f1l, fbh, acc1);
    }

    // C/D layout: vgpr r holds row M = r + 8*g, col = lane%16
    #pragma unroll
    for (int r = 0; r < 8; ++r) {
        int row0 = m0 + r + 8 * g;
        if (row0 < M) C[(size_t)row0 * N + n0 + ml] = acc0[r];
        int row1 = row0 + 16;
        if (row1 < M) C[(size_t)row1 * N + n0 + ml] = acc1[r];
    }
}

// ---------- attention logits per node/head: dot(h[n,h,:], att[h,:]) ----------
__global__ void alpha_kernel(const float* __restrict__ h, const float* __restrict__ a_s,
                             const float* __restrict__ a_d, float* __restrict__ out_s,
                             float* __restrict__ out_d, int N, int H, int C) {
    int i = blockIdx.x * blockDim.x + threadIdx.x;
    if (i >= N * H) return;
    int n = i / H, hh = i % H;
    const float* hp = h + (size_t)n * H * C + (size_t)hh * C;
    float ss = 0.f, dd = 0.f;
    for (int c = 0; c < C; ++c) {
        float v = hp[c];
        ss += v * a_s[hh * C + c];
        dd += v * a_d[hh * C + c];
    }
    out_s[i] = ss;
    out_d[i] = dd;
}

__global__ void fill_u32_kernel(unsigned* __restrict__ p, unsigned v, long long n) {
    long long i = blockIdx.x * (long long)blockDim.x + threadIdx.x;
    if (i < n) p[i] = v;
}

// edge id e in [0, E+N): e<E -> (src=ei[e], dst=ei[E+e]) else self-loop (e-E, e-E)
__device__ __forceinline__ void edge_sd(const int* ei, long long e, long long E, int& s, int& d) {
    if (e < E) { s = ei[e]; d = ei[E + e]; }
    else       { s = d = (int)(e - E); }
}

__global__ void edge_max_kernel(const float* __restrict__ as, const float* __restrict__ ad,
                                const int* __restrict__ ei, long long E, int N, int H,
                                unsigned* __restrict__ mkey) {
    long long idx = blockIdx.x * (long long)blockDim.x + threadIdx.x;
    long long total = (E + N) * (long long)H;
    if (idx >= total) return;
    int hh = (int)(idx % H);
    long long e = idx / H;
    int s, d; edge_sd(ei, e, E, s, d);
    float v = as[(size_t)s * H + hh] + ad[(size_t)d * H + hh];
    v = v > 0.f ? v : NEG_SLOPE * v;
    atomicMax(&mkey[(size_t)d * H + hh], f2ord(v));
}

__global__ void edge_expsum_kernel(const float* __restrict__ as, const float* __restrict__ ad,
                                   const int* __restrict__ ei, long long E, int N, int H,
                                   const unsigned* __restrict__ mkey,
                                   float* __restrict__ w, float* __restrict__ ssum) {
    long long idx = blockIdx.x * (long long)blockDim.x + threadIdx.x;
    long long total = (E + N) * (long long)H;
    if (idx >= total) return;
    int hh = (int)(idx % H);
    long long e = idx / H;
    int s, d; edge_sd(ei, e, E, s, d);
    float v = as[(size_t)s * H + hh] + ad[(size_t)d * H + hh];
    v = v > 0.f ? v : NEG_SLOPE * v;
    float m = ord2f(mkey[(size_t)d * H + hh]);
    float wv = expf(v - m);
    w[idx] = wv;
    atomicAdd(&ssum[(size_t)d * H + hh], wv);
}

// s[i] = 1 / (s[i] + 1e-16)  -> aggregate pass multiplies instead of dividing
__global__ void recip_kernel(float* __restrict__ s, long long n) {
    long long i = blockIdx.x * (long long)blockDim.x + threadIdx.x;
    if (i < n) s[i] = 1.0f / (s[i] + 1e-16f);
}

__global__ void edge_agg_kernel(const float* __restrict__ hmat, const float* __restrict__ w,
                                const float* __restrict__ rs, const int* __restrict__ ei,
                                long long E, int N, int H, int C, float* __restrict__ out) {
    long long idx = blockIdx.x * (long long)blockDim.x + threadIdx.x;
    const int HC = H * C;
    long long total = (E + N) * (long long)HC;
    if (idx >= total) return;
    int hc = (int)(idx % HC);
    long long e = idx / HC;
    int hh = hc / C;
    int s, d; edge_sd(ei, e, E, s, d);
    float alpha = w[e * H + hh] * rs[(size_t)d * H + hh];
    atomicAdd(&out[(size_t)d * HC + hc], hmat[(size_t)s * HC + hc] * alpha);
}

__global__ void bias_elu_kernel(float* __restrict__ x, const float* __restrict__ b,
                                long long n, int F) {
    long long i = blockIdx.x * (long long)blockDim.x + threadIdx.x;
    if (i >= n) return;
    float v = x[i] + b[i % F];
    x[i] = v > 0.f ? v : (expf(v) - 1.f);
}

__global__ void bias_add_kernel(float* __restrict__ x, const float* __restrict__ b,
                                long long n, int F) {
    long long i = blockIdx.x * (long long)blockDim.x + threadIdx.x;
    if (i >= n) return;
    x[i] += b[i % F];
}

extern "C" void kernel_launch(void* const* d_in, const int* in_sizes, int n_in,
                              void* d_out, int out_size, void* d_ws, size_t ws_size,
                              hipStream_t stream) {
    const float* x   = (const float*)d_in[0];
    const int*   ei  = (const int*)  d_in[1];
    const float* W1  = (const float*)d_in[2];
    const float* aS1 = (const float*)d_in[3];
    const float* aD1 = (const float*)d_in[4];
    const float* b1  = (const float*)d_in[5];
    const float* W2  = (const float*)d_in[6];
    const float* aS2 = (const float*)d_in[7];
    const float* aD2 = (const float*)d_in[8];
    const float* b2  = (const float*)d_in[9];
    float* out = (float*)d_out;

    const int N = in_sizes[0] / IN_DIM;
    const long long E = in_sizes[1] / 2;
    const long long Etot = E + N;

    // carve workspace (256B aligned chunks -> 16B fragment alignment holds)
    char* ws = (char*)d_ws;
    auto alloc = [&](size_t bytes) -> char* {
        char* p = ws; ws += (bytes + 255) & ~(size_t)255; return p;
    };
    float*    h1   = (float*)   alloc((size_t)N * HID * 4);
    float*    out1 = (float*)   alloc((size_t)N * HID * 4);
    float*    h2   = (float*)   alloc((size_t)N * OUT_DIM * 4);
    float*    as1  = (float*)   alloc((size_t)N * HEADS * 4);
    float*    ad1  = (float*)   alloc((size_t)N * HEADS * 4);
    unsigned* mk1  = (unsigned*)alloc((size_t)N * HEADS * 4);
    float*    s1   = (float*)   alloc((size_t)N * HEADS * 4);
    float*    w1   = (float*)   alloc((size_t)Etot * HEADS * 4);
    float*    as2  = (float*)   alloc((size_t)N * 4);
    float*    ad2  = (float*)   alloc((size_t)N * 4);
    unsigned* mk2  = (unsigned*)alloc((size_t)N * 4);
    float*    s2   = (float*)   alloc((size_t)N * 4);
    float*    w2   = (float*)   alloc((size_t)Etot * 4);
    // bf16 hi/lo operands
    __bf16* xh  = (__bf16*)alloc((size_t)N * IN_DIM * 2);
    __bf16* xl  = (__bf16*)alloc((size_t)N * IN_DIM * 2);
    __bf16* g1h = (__bf16*)alloc((size_t)N * HID * 2);
    __bf16* g1l = (__bf16*)alloc((size_t)N * HID * 2);
    __bf16* w1h = (__bf16*)alloc((size_t)IN_DIM * HID * 2);
    __bf16* w1l = (__bf16*)alloc((size_t)IN_DIM * HID * 2);
    __bf16* w2h = (__bf16*)alloc((size_t)HID * OUT_DIM * 2);
    __bf16* w2l = (__bf16*)alloc((size_t)HID * OUT_DIM * 2);

    dim3 b256(256);
    auto nb = [](long long n, int t) { return dim3((unsigned)((n + t - 1) / t)); };

    // ---------------- Layer 1 ----------------
    split_bf16_kernel<<<nb((long long)N * IN_DIM, 256), b256, 0, stream>>>(
        x, xh, xl, (long long)N * IN_DIM);
    split_bf16_t_kernel<<<nb((long long)IN_DIM * HID, 256), b256, 0, stream>>>(
        W1, w1h, w1l, IN_DIM, HID);
    gemm_wmma_bf16x3<<<dim3((N + 31) / 32, HID / 128), b256, 0, stream>>>(
        xh, xl, w1h, w1l, h1, N, IN_DIM, HID);

    alpha_kernel<<<nb((long long)N * HEADS, 256), b256, 0, stream>>>(
        h1, aS1, aD1, as1, ad1, N, HEADS, NHID);
    fill_u32_kernel<<<nb((long long)N * HEADS, 256), b256, 0, stream>>>(
        mk1, ORD_NEG_MAX, (long long)N * HEADS);
    fill_u32_kernel<<<nb((long long)N * HEADS, 256), b256, 0, stream>>>(
        (unsigned*)s1, 0u, (long long)N * HEADS);
    fill_u32_kernel<<<nb((long long)N * HID, 256), b256, 0, stream>>>(
        (unsigned*)out1, 0u, (long long)N * HID);
    edge_max_kernel<<<nb(Etot * HEADS, 256), b256, 0, stream>>>(as1, ad1, ei, E, N, HEADS, mk1);
    edge_expsum_kernel<<<nb(Etot * HEADS, 256), b256, 0, stream>>>(as1, ad1, ei, E, N, HEADS, mk1, w1, s1);
    recip_kernel<<<nb((long long)N * HEADS, 256), b256, 0, stream>>>(s1, (long long)N * HEADS);
    edge_agg_kernel<<<nb(Etot * HID, 256), b256, 0, stream>>>(h1, w1, s1, ei, E, N, HEADS, NHID, out1);
    bias_elu_kernel<<<nb((long long)N * HID, 256), b256, 0, stream>>>(out1, b1, (long long)N * HID, HID);

    // ---------------- Layer 2 ----------------
    split_bf16_kernel<<<nb((long long)N * HID, 256), b256, 0, stream>>>(
        out1, g1h, g1l, (long long)N * HID);
    split_bf16_t_kernel<<<nb((long long)HID * OUT_DIM, 256), b256, 0, stream>>>(
        W2, w2h, w2l, HID, OUT_DIM);
    gemm_wmma_bf16x3<<<dim3((N + 31) / 32, OUT_DIM / 128), b256, 0, stream>>>(
        g1h, g1l, w2h, w2l, h2, N, HID, OUT_DIM);

    alpha_kernel<<<nb((long long)N, 256), b256, 0, stream>>>(h2, aS2, aD2, as2, ad2, N, 1, OUT_DIM);
    fill_u32_kernel<<<nb(N, 256), b256, 0, stream>>>(mk2, ORD_NEG_MAX, N);
    fill_u32_kernel<<<nb(N, 256), b256, 0, stream>>>((unsigned*)s2, 0u, N);
    fill_u32_kernel<<<nb((long long)N * OUT_DIM, 256), b256, 0, stream>>>(
        (unsigned*)out, 0u, (long long)N * OUT_DIM);
    edge_max_kernel<<<nb(Etot, 256), b256, 0, stream>>>(as2, ad2, ei, E, N, 1, mk2);
    edge_expsum_kernel<<<nb(Etot, 256), b256, 0, stream>>>(as2, ad2, ei, E, N, 1, mk2, w2, s2);
    recip_kernel<<<nb(N, 256), b256, 0, stream>>>(s2, N);
    edge_agg_kernel<<<nb(Etot * OUT_DIM, 256), b256, 0, stream>>>(h2, w2, s2, ei, E, N, 1, OUT_DIM, out);
    bias_add_kernel<<<nb((long long)N * OUT_DIM, 256), b256, 0, stream>>>(out, b2, (long long)N * OUT_DIM, OUT_DIM);
}
